// GraphAttentionLayerCentroids_81183471829341
// MI455X (gfx1250) — compile-verified, matrix operand order checked by reading, not run on hardware
//
#include <hip/hip_runtime.h>
#include <hip/hip_bf16.h>
#include <stdint.h>

typedef __attribute__((ext_vector_type(16))) _Float16 v16h;
typedef __attribute__((ext_vector_type(8)))  _Float16 v8h;
typedef __attribute__((ext_vector_type(8)))  float    v8f;

#define B_  64
#define C_  1024
#define N_  512
#define H_  8
#define D_  128
#define LOG2E 1.44269504088896340736f

static __device__ __forceinline__ v16h cat16(v8h lo, v8h hi8) {
    return __builtin_shufflevector(lo, hi8, 0,1,2,3,4,5,6,7,8,9,10,11,12,13,14,15);
}

static __device__ __forceinline__ v8f wmma_f16(v16h a, v16h b, v8f c) {
    return __builtin_amdgcn_wmma_f32_16x16x32_f16(
        /*neg_a=*/false, a, /*neg_b=*/false, b,
        /*c_mod=*/(short)0, c, /*reuse_a=*/false, /*reuse_b=*/false);
}

// ---------------------------------------------------------------------------
// Kernel 0: convert W_w (fp32, [C,C] row-major) to f16 in workspace
// ---------------------------------------------------------------------------
__global__ __launch_bounds__(256) void cvt_w_kernel(const float* __restrict__ W,
                                                    _Float16* __restrict__ Wh) {
    int i = blockIdx.x * 256 + threadIdx.x;   // grid covers C_*C_ = 1M
    Wh[i] = (_Float16)W[i];
}

// ---------------------------------------------------------------------------
// Kernel 1: Xp[b,n,d] = sum_c h[b,c,n] * W[d,c] + bias[d]   (f16 out, f32 acc)
// One block = (b, 64-row n panel).  8 waves x 8 d-tiles; each B (W) fragment
// is reused across 4 n-tiles -> 4x less W traffic from L2.
// ---------------------------------------------------------------------------
#define ASTRIDE 1040   // 1024 + 16 halves pad; 2080 B row stride (16B aligned)
__global__ __launch_bounds__(256) void proj_kernel(const float* __restrict__ h,
                                                   const _Float16* __restrict__ Wh,
                                                   const float* __restrict__ Wb,
                                                   _Float16* __restrict__ Xp) {
    __shared__ __align__(16) _Float16 As[64 * ASTRIDE];   // 133 KB

    const int blk = blockIdx.x;          // B_ * (N_/64) = 512
    const int b  = blk >> 3;
    const int n0 = (blk & 7) * 64;
    const int t  = threadIdx.x;

    // Cooperative transpose-load: As[nn][c] = (f16) h[b, c, n0+nn]
    {
        const int nn = t & 63;
        const int cb = t >> 6;           // 0..3
        const float* hb = h + (size_t)b * C_ * N_;
        #pragma unroll 8
        for (int i = 0; i < 256; ++i) {
            int c = cb + 4 * i;
            As[nn * ASTRIDE + c] = (_Float16)hb[(size_t)c * N_ + n0 + nn];
        }
    }
    __syncthreads();

    const int w    = t >> 5;
    const int lane = t & 31;
    const int m    = lane & 15;
    const int hi   = lane >> 4;

    #pragma unroll 1
    for (int dt = 0; dt < 8; ++dt) {
        const int d0   = (w * 8 + dt) * 16;
        const int dcol = d0 + m;                    // output column for this lane
        const float bias = Wb[dcol];
        v8f acc[4];
        #pragma unroll
        for (int nt = 0; nt < 4; ++nt)
            acc[nt] = (v8f){bias, bias, bias, bias, bias, bias, bias, bias};

        #pragma unroll 2
        for (int k0 = 0; k0 < C_; k0 += 32) {
            const _Float16* wp = Wh + (size_t)dcol * C_ + k0 + hi * 8;
            v16h bb = cat16(*(const v8h*)wp, *(const v8h*)(wp + 16));
            #pragma unroll
            for (int nt = 0; nt < 4; ++nt) {
                const _Float16* ap = &As[(nt * 16 + m) * ASTRIDE + k0 + hi * 8];
                v16h aa = cat16(*(const v8h*)ap, *(const v8h*)(ap + 16));
                acc[nt] = wmma_f16(aa, bb, acc[nt]);
            }
        }

        #pragma unroll
        for (int nt = 0; nt < 4; ++nt) {
            _Float16* yp = Xp + ((size_t)b * N_ + n0 + nt * 16) * C_ + dcol;
            #pragma unroll
            for (int r = 0; r < 8; ++r)
                yp[(size_t)(r + 8 * hi) * C_] = (_Float16)acc[nt][r];
        }
    }
}

// ---------------------------------------------------------------------------
// Kernel 2: per-head unscaled self-attention with online softmax.
// Q = K = V = Xp[b, :, h*128 : h*128+128].  One block = (b, h, 128-query blk).
// K/V panels double-buffered in LDS via GLOBAL_LOAD_ASYNC_TO_LDS_B128:
// panel kb+1 streams in (ASYNCcnt) while the 64 WMMAs of panel kb execute.
// ---------------------------------------------------------------------------
#define KB_STRIDE 136    // 128 + 8 halves pad
__global__ __launch_bounds__(256) void attn_kernel(const _Float16* __restrict__ Xp,
                                                   _Float16* __restrict__ Y) {
    __shared__ __align__(16) _Float16 Kblk[2][128 * KB_STRIDE];  // 2 x 34 KB
    __shared__ __align__(16) _Float16 Pbuf[8][16 * 32];          // 8 KB

    const int blk = blockIdx.x;          // B_ * H_ * 4 = 2048
    const int qb  = blk & 3;
    const int hh  = (blk >> 2) & 7;
    const int b   = blk >> 5;
    const int t   = threadIdx.x;
    const int w    = t >> 5;
    const int lane = t & 31;
    const int m    = lane & 15;
    const int hi   = lane >> 4;
    const int q0   = qb * 128 + w * 16;

    const _Float16* Xhd = Xp + (size_t)b * N_ * C_ + hh * D_;

    // Issue this thread's share of async global->LDS copies for one panel
    auto stage = [&](int kb, int buf) {
        const int row = t >> 1;
        const int hs  = t & 1;
        const _Float16* kp = Xhd + (size_t)(kb * 128 + row) * C_ + hs * 64;
        const uint32_t dls =
            (uint32_t)(uintptr_t)&Kblk[buf][row * KB_STRIDE + hs * 64];
        #pragma unroll
        for (int i = 0; i < 8; ++i) {
            uint64_t ga = (uint64_t)(uintptr_t)(kp + i * 8);
            uint32_t la = dls + i * 16;
            asm volatile("global_load_async_to_lds_b128 %0, %1, off"
                         :: "v"(la), "v"(ga) : "memory");
        }
    };

    // Preload this wave's 16 Q rows as A fragments (4 K-steps of 32)
    v16h qa[4];
    {
        const _Float16* qp = Xhd + (size_t)(q0 + m) * C_;
        #pragma unroll
        for (int kk = 0; kk < 4; ++kk) {
            v8h lo  = *(const v8h*)(qp + kk * 32 + hi * 8);
            v8h hi8 = *(const v8h*)(qp + kk * 32 + 16 + hi * 8);
            qa[kk] = cat16(lo, hi8);
        }
    }

    v8f acc[8];
    #pragma unroll
    for (int i = 0; i < 8; ++i) acc[i] = (v8f){0,0,0,0,0,0,0,0};
    float mrow[8], lrow[8];
    #pragma unroll
    for (int r = 0; r < 8; ++r) { mrow[r] = -__builtin_inff(); lrow[r] = 0.f; }

    // Prologue: fill panel 0, drain, make visible block-wide
    stage(0, 0);
    asm volatile("s_wait_asynccnt 0" ::: "memory");
    __syncthreads();

    for (int kb = 0; kb < 4; ++kb) {
        // Kick off next panel into the other buffer; do NOT wait yet.
        if (kb < 3) stage(kb + 1, (kb + 1) & 1);
        const _Float16* Kb = &Kblk[kb & 1][0];

        for (int jj = 0; jj < 4; ++jj) {      // 32 keys per inner iteration
            const int krow0 = jj * 32;
            // S = Q * K^T : two 16x16 C-tiles (keys krow0.. and krow0+16..)
            v8f s0 = {0,0,0,0,0,0,0,0}, s1 = {0,0,0,0,0,0,0,0};
            #pragma unroll
            for (int kk = 0; kk < 4; ++kk) {
                const _Float16* bp0 = &Kb[(krow0 + m) * KB_STRIDE + kk * 32 + hi * 8];
                s0 = wmma_f16(qa[kk], cat16(*(const v8h*)bp0, *(const v8h*)(bp0 + 16)), s0);
                const _Float16* bp1 = &Kb[(krow0 + 16 + m) * KB_STRIDE + kk * 32 + hi * 8];
                s1 = wmma_f16(qa[kk], cat16(*(const v8h*)bp1, *(const v8h*)(bp1 + 16)), s1);
            }

            // Online-softmax update; row stats shared across the 16-lane half
            float p0v[8], p1v[8];
            #pragma unroll
            for (int r = 0; r < 8; ++r) {
                float mx = fmaxf(s0[r], s1[r]);
                mx = fmaxf(mx, __shfl_xor(mx, 1, 32));
                mx = fmaxf(mx, __shfl_xor(mx, 2, 32));
                mx = fmaxf(mx, __shfl_xor(mx, 4, 32));
                mx = fmaxf(mx, __shfl_xor(mx, 8, 32));
                float mnew  = fmaxf(mrow[r], mx);
                float scale = exp2f((mrow[r] - mnew) * LOG2E);
                mrow[r] = mnew;
                float p0 = exp2f((s0[r] - mnew) * LOG2E);
                float p1 = exp2f((s1[r] - mnew) * LOG2E);
                float rs = p0 + p1;
                rs += __shfl_xor(rs, 1, 32);
                rs += __shfl_xor(rs, 2, 32);
                rs += __shfl_xor(rs, 4, 32);
                rs += __shfl_xor(rs, 8, 32);
                lrow[r] = lrow[r] * scale + rs;
                p0v[r] = p0; p1v[r] = p1;
                #pragma unroll
                for (int dt = 0; dt < 8; ++dt) acc[dt][r] *= scale;
            }

            // C-layout -> A-layout for P via per-wave LDS bounce
            _Float16* pb = &Pbuf[w][0];
            #pragma unroll
            for (int r = 0; r < 8; ++r) {
                pb[(r + 8 * hi) * 32 + m]      = (_Float16)p0v[r];
                pb[(r + 8 * hi) * 32 + 16 + m] = (_Float16)p1v[r];
            }
            asm volatile("s_wait_dscnt 0" ::: "memory");
            v16h pa = cat16(*(const v8h*)(pb + m * 32 + hi * 8),
                            *(const v8h*)(pb + m * 32 + 16 + hi * 8));

            // O += P * V   (V rows are the same LDS panel, read column-wise)
            #pragma unroll
            for (int dt = 0; dt < 8; ++dt) {
                const int d = dt * 16 + m;
                v16h bv;
                #pragma unroll
                for (int j = 0; j < 8; ++j) {
                    bv[j]     = Kb[(krow0 + hi * 8 + j) * KB_STRIDE + d];
                    bv[j + 8] = Kb[(krow0 + 16 + hi * 8 + j) * KB_STRIDE + d];
                }
                acc[dt] = wmma_f16(pa, bv, acc[dt]);
            }
        }

        // Drain this wave's in-flight async fills of the next panel, then
        // rendezvous: after the barrier, panel kb+1 is complete block-wide
        // and nobody still reads the buffer that iteration kb+1 will refill.
        asm volatile("s_wait_asynccnt 0" ::: "memory");
        __syncthreads();
    }

    // Normalize and store f16 head output
    float inv[8];
    #pragma unroll
    for (int r = 0; r < 8; ++r) inv[r] = 1.0f / lrow[r];
    #pragma unroll
    for (int dt = 0; dt < 8; ++dt) {
        #pragma unroll
        for (int r = 0; r < 8; ++r) {
            size_t idx = ((size_t)b * N_ + q0 + r + 8 * hi) * C_ + hh * D_ + dt * 16 + m;
            Y[idx] = (_Float16)(acc[dt][r] * inv[r]);
        }
    }
}

// ---------------------------------------------------------------------------
// Kernel 3: LayerNorm(C) + exact-erf GELU + transpose to out[b,c,n] (fp32)
// One block = (b, 16-column n tile).  LDS bounce so the transposed stores are
// coalesced (lanes 0..15 cover 16 consecutive n -> 64B segments).
// ---------------------------------------------------------------------------
#define YB_STRIDE 1032
__global__ __launch_bounds__(256) void ln_gelu_kernel(const _Float16* __restrict__ Y,
                                                      const float* __restrict__ g,
                                                      const float* __restrict__ be,
                                                      float* __restrict__ out) {
    __shared__ __align__(16) _Float16 Ybuf[16 * YB_STRIDE];   // 33 KB
    __shared__ float smu[16], srs[16];

    const int blk = blockIdx.x;          // B_ * (N_/16) = 2048
    const int b  = blk >> 5;
    const int n0 = (blk & 31) * 16;
    const int t  = threadIdx.x;

    // Coalesced f16 load of Y[b, n0..n0+16, :]
    {
        const int row = t >> 4;          // 0..15
        const int ch  = t & 15;
        const _Float16* yp = Y + ((size_t)b * N_ + n0 + row) * C_;
        #pragma unroll
        for (int i = 0; i < 8; ++i) {
            int c8 = (ch + 16 * i) * 8;
            *(v8h*)&Ybuf[row * YB_STRIDE + c8] = *(const v8h*)(yp + c8);
        }
    }
    __syncthreads();

    // Per-row mean / rstd: 16 lanes per row, 64 channels each, shfl-reduce
    {
        const int row  = t >> 4;
        const int part = t & 15;
        float s = 0.f, s2 = 0.f;
        #pragma unroll 8
        for (int i = 0; i < 64; ++i) {
            float x = (float)Ybuf[row * YB_STRIDE + part * 64 + i];
            s += x; s2 += x * x;
        }
        s  += __shfl_xor(s, 1, 32);  s2 += __shfl_xor(s2, 1, 32);
        s  += __shfl_xor(s, 2, 32);  s2 += __shfl_xor(s2, 2, 32);
        s  += __shfl_xor(s, 4, 32);  s2 += __shfl_xor(s2, 4, 32);
        s  += __shfl_xor(s, 8, 32);  s2 += __shfl_xor(s2, 8, 32);
        if (part == 0) {
            float mu  = s * (1.0f / C_);
            float var = s2 * (1.0f / C_) - mu * mu;
            smu[row] = mu;
            srs[row] = rsqrtf(var + 1e-5f);
        }
    }
    __syncthreads();

    // Transposed, coalesced store: lane group shares c, writes 16 adjacent n
    {
        const int nn = t & 15;           // n offset == Ybuf row
        const int cb = t >> 4;           // 0..15
        const float mu = smu[nn];
        const float rs = srs[nn];
        const size_t ob = (size_t)b * C_ * N_ + n0 + nn;
        #pragma unroll 4
        for (int i = 0; i < 64; ++i) {
            int c = cb + 16 * i;
            float x  = ((float)Ybuf[nn * YB_STRIDE + c] - mu) * rs * g[c] + be[c];
            float ge = 0.5f * x * (1.0f + erff(x * 0.70710678118654752f));
            out[ob + (size_t)c * N_] = ge;
        }
    }
}

// ---------------------------------------------------------------------------
extern "C" void kernel_launch(void* const* d_in, const int* in_sizes, int n_in,
                              void* d_out, int out_size, void* d_ws, size_t ws_size,
                              hipStream_t stream) {
    (void)in_sizes; (void)n_in; (void)out_size; (void)ws_size;
    // setup_inputs order: t, h, W_w, W_b, ln_g, ln_b
    const float* h  = (const float*)d_in[1];
    const float* Ww = (const float*)d_in[2];
    const float* Wb = (const float*)d_in[3];
    const float* g  = (const float*)d_in[4];
    const float* be = (const float*)d_in[5];
    float* out = (float*)d_out;

    char* ws = (char*)d_ws;
    _Float16* Wh = (_Float16*)ws;                                   // 2 MB
    _Float16* Xp = (_Float16*)(ws + ((size_t)2  << 20));            // 64 MB
    _Float16* Yb = (_Float16*)(ws + ((size_t)66 << 20));            // 64 MB

    cvt_w_kernel<<<(C_ * C_) / 256, 256, 0, stream>>>(Ww, Wh);
    proj_kernel<<<B_ * (N_ / 64), 256, 0, stream>>>(h, Wh, Wb, Xp);
    attn_kernel<<<B_ * H_ * (N_ / 128), 256, 0, stream>>>(Xp, Yb);
    ln_gelu_kernel<<<B_ * (N_ / 16), 256, 0, stream>>>(Yb, g, be, out);
}
